// EucCluster_10694468567400
// MI455X (gfx1250) — matrix operand order repaired and checked by reading, not compiled, not run on hardware
//
#include <hip/hip_runtime.h>
#include <hip/hip_bf16.h>

// ---------------------------------------------------------------------------
// EucCluster for MI455X (gfx1250, wave32, WMMA).
//
// Compute-bound: 34.4 GFLOP cross-term GEMM vs ~70 MB HBM traffic
// (~500 FLOP/B). Structure:
//  - x-tile staged f32->f16 in LDS (padded stride, conflict-free b128),
//    then hoisted to registers: 2 row-stripes x full-K A = 128 VGPRs/wave.
//  - Each 32B/lane B fragment (layout-exact for v_wmma_f32_16x16x32_f16)
//    feeds TWO wmmas (register-blocked M) -> B operand refill halved.
//  - v3: B fragment software-pipelined one K-step ahead so the compiler
//    can issue loads early and use partial s_wait_loadcnt instead of
//    draining to 0 before every wmma pair.
//  - Precision: f16 only for x.c (err ~8e-5 << inter-center spread ~0.16);
//    ||x||^2, ||c||^2, combine, clamp, sqrt, reductions all f32.
//  - Argmin: monotone packed (f32bits<<32 | idx) u64 keys; ds_min_u64 per
//    block then one global_atomic_min_u64 per center per block (ties ->
//    smallest index, matching jnp.argmin first-occurrence).
// ---------------------------------------------------------------------------

#define N_PTS  65536
#define M_CTR  1024
#define D_DIM  256
#define BM     256        // points per block (8 waves x 2 stripes x 16 rows)
#define LDA    264        // padded LDS row stride in halfs (+16B kills bank conflicts)

typedef __attribute__((ext_vector_type(16))) _Float16 v16h;
typedef __attribute__((ext_vector_type(8)))  _Float16 v8h;
typedef __attribute__((ext_vector_type(8)))  float    v8f;

__device__ __forceinline__ v16h load_a_frag(const _Float16* p) {
    v8h lo = *(const v8h*)(p);        // K = kbase .. kbase+7
    v8h hi = *(const v8h*)(p + 16);   // K = kbase+16 .. kbase+23
    return __builtin_shufflevector(lo, hi,
            0, 1, 2, 3, 4, 5, 6, 7, 8, 9, 10, 11, 12, 13, 14, 15);
}

// --------------------------- kernel 1: prep centers -------------------------
__global__ __launch_bounds__(256)
void euc_prep_centers(const float* __restrict__ centers,
                      _Float16* __restrict__ cH,
                      float* __restrict__ c2,
                      unsigned long long* __restrict__ packed) {
    const int wave = threadIdx.x >> 5;
    const int lane = threadIdx.x & 31;
    const int row  = blockIdx.x * 8 + wave;          // 128 blocks * 8 waves = 1024 rows

    const float4* src = (const float4*)(centers + (size_t)row * D_DIM + lane * 8);
    float4 a = src[0];
    float4 b = src[1];
    float s = a.x * a.x + a.y * a.y + a.z * a.z + a.w * a.w
            + b.x * b.x + b.y * b.y + b.z * b.z + b.w * b.w;

    v8h h;
    h[0] = (_Float16)a.x; h[1] = (_Float16)a.y; h[2] = (_Float16)a.z; h[3] = (_Float16)a.w;
    h[4] = (_Float16)b.x; h[5] = (_Float16)b.y; h[6] = (_Float16)b.z; h[7] = (_Float16)b.w;
    *(v8h*)(cH + (size_t)row * D_DIM + lane * 8) = h;

    #pragma unroll
    for (int off = 16; off; off >>= 1) s += __shfl_xor(s, off);
    if (lane == 0) c2[row] = s;

    const int gid = blockIdx.x * 256 + threadIdx.x;
    if (gid < M_CTR) packed[gid] = ~0ull;            // +inf key
}

// --------------------------- kernel 2: main fused GEMM+reduce ---------------
__global__ __launch_bounds__(256)
void euc_main(const float* __restrict__ x,
              const _Float16* __restrict__ cH,
              const float* __restrict__ c2g,
              unsigned long long* __restrict__ packedG,
              float* __restrict__ minOut) {
    __shared__ __attribute__((aligned(16))) _Float16 sA[BM * LDA];   // 132 KB f16 x-tile
    __shared__ float sX2[BM];
    __shared__ float sC2[M_CTR];
    __shared__ unsigned long long sCol[M_CTR];                       // per-block argmin keys

    const int tid  = threadIdx.x;
    const int wave = tid >> 5;
    const int lane = tid & 31;
    const int rowBlock = blockIdx.x * BM;

    for (int i = tid; i < M_CTR; i += 256) {
        sCol[i] = ~0ull;
        sC2[i]  = c2g[i];
    }

    // Stage x-tile (f32 -> f16 in LDS) and compute ||x||^2 in f32.
    for (int r = wave; r < BM; r += 8) {
        const float4* src = (const float4*)(x + (size_t)(rowBlock + r) * D_DIM + lane * 8);
        float4 a = src[0];
        float4 b = src[1];
        float s = a.x * a.x + a.y * a.y + a.z * a.z + a.w * a.w
                + b.x * b.x + b.y * b.y + b.z * b.z + b.w * b.w;
        v8h h;
        h[0] = (_Float16)a.x; h[1] = (_Float16)a.y; h[2] = (_Float16)a.z; h[3] = (_Float16)a.w;
        h[4] = (_Float16)b.x; h[5] = (_Float16)b.y; h[6] = (_Float16)b.z; h[7] = (_Float16)b.w;
        *(v8h*)&sA[r * LDA + lane * 8] = h;
        #pragma unroll
        for (int off = 16; off; off >>= 1) s += __shfl_xor(s, off);
        if (lane == 0) sX2[r] = s;
    }
    __syncthreads();

    const int hi   = lane >> 4;      // 0: rows j / K 0..7|16..23; 1: rows 8+j / K 8..15|24..31
    const int lcol = lane & 15;      // column within the 16-wide tile
    const int wrow = wave * 32;      // this wave's 32-row super-stripe (2 x 16)

    // Preload both A stripes for the full K range: 2 x 8 x v16h = 128 VGPRs.
    v16h aReg0[8], aReg1[8];
    {
        const _Float16* aBase0 = &sA[(wrow +      lcol) * LDA + hi * 8];
        const _Float16* aBase1 = &sA[(wrow + 16 + lcol) * LDA + hi * 8];
        #pragma unroll
        for (int kk = 0; kk < 8; kk++) {
            aReg0[kk] = load_a_frag(aBase0 + kk * 32);
            aReg1[kk] = load_a_frag(aBase1 + kk * 32);
        }
    }

    float x2v0[8], x2v1[8], rowMin0[8], rowMin1[8];
    #pragma unroll
    for (int j = 0; j < 8; j++) {
        x2v0[j] = sX2[wrow +      hi * 8 + j];
        x2v1[j] = sX2[wrow + 16 + hi * 8 + j];
        rowMin0[j] = 3.402823466e38f;
        rowMin1[j] = 3.402823466e38f;
    }

    for (int n0 = 0; n0 < M_CTR; n0 += 16) {
        v8f acc0 = {};
        v8f acc1 = {};
        // B fragment: lane = column, K = 8h..8h+7 | 16+8h..16+8h+7 per lane.
        const _Float16* bPtr = cH + (size_t)(n0 + lcol) * D_DIM + hi * 8;
        if (n0 + 16 < M_CTR)
            __builtin_prefetch(bPtr + 16 * D_DIM, 0, 1);   // warm next column tile

        // Software-pipelined K loop: fetch fragment kk+1 while wmma'ing kk,
        // so waits become partial (loadcnt<=2) instead of full drains.
        v16h bcur = load_a_frag(bPtr);
        #pragma unroll
        for (int kk = 0; kk < 7; kk++) {
            v16h bnext = load_a_frag(bPtr + (kk + 1) * 32);
            acc0 = __builtin_amdgcn_wmma_f32_16x16x32_f16(
                       false, aReg0[kk], false, bcur, (short)0, acc0, false, false);
            acc1 = __builtin_amdgcn_wmma_f32_16x16x32_f16(
                       false, aReg1[kk], false, bcur, (short)0, acc1, false, false);
            bcur = bnext;
        }
        acc0 = __builtin_amdgcn_wmma_f32_16x16x32_f16(
                   false, aReg0[7], false, bcur, (short)0, acc0, false, false);
        acc1 = __builtin_amdgcn_wmma_f32_16x16x32_f16(
                   false, aReg1[7], false, bcur, (short)0, acc1, false, false);

        const float c2c = sC2[n0 + lcol];
        unsigned long long best = ~0ull;
        #pragma unroll
        for (int j = 0; j < 8; j++) {
            float d2a = fmaxf(x2v0[j] + c2c - 2.0f * acc0[j], 0.0f);
            float d2b = fmaxf(x2v1[j] + c2c - 2.0f * acc1[j], 0.0f);
            rowMin0[j] = fminf(rowMin0[j], d2a);
            rowMin1[j] = fminf(rowMin1[j], d2b);
            unsigned long long ka =
                ((unsigned long long)__float_as_uint(d2a) << 32) |
                (unsigned long long)(rowBlock + wrow + hi * 8 + j);
            unsigned long long kb =
                ((unsigned long long)__float_as_uint(d2b) << 32) |
                (unsigned long long)(rowBlock + wrow + 16 + hi * 8 + j);
            best = (ka < best) ? ka : best;
            best = (kb < best) ? kb : best;
        }
        // Combine the two half-wave row groups for this column (lane ^ 16).
        {
            unsigned int bh = (unsigned int)(best >> 32);
            unsigned int bl = (unsigned int)best;
            unsigned int oh = __shfl_xor(bh, 16);
            unsigned int ol = __shfl_xor(bl, 16);
            unsigned long long other = ((unsigned long long)oh << 32) | ol;
            if (other < best) best = other;
        }
        if (hi == 0)
            atomicMin(&sCol[n0 + lcol], best);          // ds_min_u64
    }

    // Per-point min distance: reduce across the 16 columns held by each half-wave.
    #pragma unroll
    for (int off = 8; off; off >>= 1) {
        #pragma unroll
        for (int j = 0; j < 8; j++) {
            rowMin0[j] = fminf(rowMin0[j], __shfl_xor(rowMin0[j], off));
            rowMin1[j] = fminf(rowMin1[j], __shfl_xor(rowMin1[j], off));
        }
    }
    if (lcol == 0) {
        #pragma unroll
        for (int j = 0; j < 8; j++) {
            minOut[rowBlock + wrow +      hi * 8 + j] = sqrtf(rowMin0[j]);
            minOut[rowBlock + wrow + 16 + hi * 8 + j] = sqrtf(rowMin1[j]);
        }
    }

    __syncthreads();
    // One global u64 atomic-min per center per block.
    for (int i = tid; i < M_CTR; i += 256)
        atomicMin(packedG + i, sCol[i]);               // global_atomic_min_u64
}

// --------------------------- kernel 3: finalize -----------------------------
__global__ __launch_bounds__(256)
void euc_finalize(const unsigned long long* __restrict__ packed,
                  const float* __restrict__ centers,
                  float* __restrict__ outArg,
                  float* __restrict__ outCenters) {
    const int gid = blockIdx.x * 256 + threadIdx.x;    // 262144 threads
    if (gid < M_CTR)
        outArg[gid] = (float)(unsigned int)(packed[gid] & 0xffffffffull);
    if (gid < M_CTR * D_DIM)
        outCenters[gid] = centers[gid];
}

// --------------------------- launch ----------------------------------------
extern "C" void kernel_launch(void* const* d_in, const int* in_sizes, int n_in,
                              void* d_out, int out_size, void* d_ws, size_t ws_size,
                              hipStream_t stream) {
    (void)in_sizes; (void)n_in; (void)out_size; (void)ws_size;
    const float* x       = (const float*)d_in[0];   // [N, D] f32
    const float* centers = (const float*)d_in[1];   // [M, D] f32

    char* ws = (char*)d_ws;
    _Float16*           cH     = (_Float16*)ws;                              // 512 KB
    float*              c2     = (float*)(ws + (size_t)512 * 1024);          //   4 KB
    unsigned long long* packed = (unsigned long long*)(ws + (size_t)512 * 1024 + 4096); // 8 KB

    float* out     = (float*)d_out;
    float* outArg  = out;                          // [M]   argmin index (as float)
    float* outMin  = out + M_CTR;                  // [N]   min distance per point
    float* outCtr  = out + M_CTR + N_PTS;          // [M*D] centers passthrough

    euc_prep_centers<<<M_CTR / 8, 256, 0, stream>>>(centers, cH, c2, packed);
    euc_main<<<N_PTS / BM, 256, 0, stream>>>(x, cH, c2, packed, outMin);
    euc_finalize<<<(M_CTR * D_DIM) / 256, 256, 0, stream>>>(packed, centers, outArg, outCtr);
}
